// SelfAttention_7387343749818
// MI455X (gfx1250) — matrix operand order repaired and checked by reading, not compiled, not run on hardware
//
#include <hip/hip_runtime.h>
#include <hip/hip_bf16.h>
#include <math.h>
#include <cstdint>

// ---------------------------------------------------------------------------
// Self-attention forward for MI455X (gfx1250): bf16 WMMA, f32 accumulate,
// async global->LDS staging (ASYNCcnt) for the A, K and V tiles.
// ---------------------------------------------------------------------------

typedef __attribute__((ext_vector_type(16))) __bf16 v16bf;
typedef __attribute__((ext_vector_type(8)))  __bf16 v8bf;
typedef __attribute__((ext_vector_type(8)))  float  v8f;
typedef __attribute__((__vector_size__(16))) int    v4i;   // matches builtin pointee

union F16x2 { v16bf v; v8bf h[2]; };

#define WMMA_BF16(a, b, c) \
  __builtin_amdgcn_wmma_f32_16x16x32_bf16(false, (a), false, (b), (short)0, (c), false, false)

// ---- CDNA5 async global->LDS copy (16B per lane), ASYNCcnt-tracked --------
__device__ __forceinline__ void async_cp16(void* lds, const void* g) {
#if __has_builtin(__builtin_amdgcn_global_load_async_to_lds_b128)
  typedef __attribute__((address_space(1))) v4i* gp_t;   // global int4*
  typedef __attribute__((address_space(3))) v4i* lp_t;   // LDS int4*
  __builtin_amdgcn_global_load_async_to_lds_b128(
      (gp_t)(uintptr_t)g,
      (lp_t)(unsigned int)(uintptr_t)lds,   // LDS offset = low 32 bits
      0, 0);
#else
  unsigned int l = (unsigned int)(uintptr_t)lds;
  unsigned long long ga = (unsigned long long)(uintptr_t)g;
  asm volatile("global_load_async_to_lds_b128 %0, %1, off" :: "v"(l), "v"(ga) : "memory");
#endif
}

#if __has_builtin(__builtin_amdgcn_s_wait_asynccnt)
#define WAIT_ASYNC(n) __builtin_amdgcn_s_wait_asynccnt(n)
#else
#define WAIT_ASYNC(n) asm volatile("s_wait_asynccnt %0" :: "i"(n) : "memory")
#endif

constexpr int BATCH = 4;
constexpr int SEQ   = 2048;
constexpr int DM    = 768;
constexpr int NH    = 8;
constexpr int DH    = 96;
constexpr int M_ROWS = BATCH * SEQ;   // 8192
constexpr int QKV_N  = 3 * DM;        // 2304

// ---------------------------------------------------------------------------
// Prologue: f32 -> bf16 (x is converted once so GEMM A-tiles can be staged
// with pure async byte copies).
// ---------------------------------------------------------------------------
__global__ __launch_bounds__(256)
void cvt_bf16(const float* __restrict__ src, __bf16* __restrict__ dst, int n8)
{
  const int i = blockIdx.x * 256 + threadIdx.x;
  if (i < n8) {
    const float4* s = (const float4*)src + (size_t)i * 2;
    float4 a = s[0], b = s[1];
    v8bf o = {(__bf16)a.x, (__bf16)a.y, (__bf16)a.z, (__bf16)a.w,
              (__bf16)b.x, (__bf16)b.y, (__bf16)b.z, (__bf16)b.w};
    *(v8bf*)(dst + (size_t)i * 8) = o;
  }
}

// ---------------------------------------------------------------------------
// Tiled GEMM:  C[M x Nsize] = A[M x Ksize](bf16) * W[Ksize x Nsize](f32) + bias
// MODE 0: scatter bf16 into Q[BH][N][Dh], K[BH][N][Dh], Vt[BH][Dh][N]
// MODE 1: f32 -> outF
// Block: 256 threads = 8 waves; 128x128x32 tiles; A staged via async copies.
// ---------------------------------------------------------------------------
template<int MODE>
__global__ __launch_bounds__(256)
void gemm_wmma(const __bf16* __restrict__ Abf, const float* __restrict__ W,
               const float* __restrict__ bias, float* __restrict__ outF,
               __bf16* __restrict__ qws, __bf16* __restrict__ kws,
               __bf16* __restrict__ vtws, int Ksize, int Nsize)
{
  __shared__ __align__(16) __bf16 As[128 * 40];   // [row][k], stride 40 (pad)
  __shared__ __align__(16) __bf16 Bs[128 * 40];   // transposed: [n][k]

  const int tid  = threadIdx.x;
  const int n0   = blockIdx.x * 128;
  const int m0   = blockIdx.y * 128;
  const int wave = tid >> 5, lane = tid & 31;
  const int half = lane >> 4, ln = lane & 15;
  const int wm0  = (wave >> 1) * 32;
  const int wn0  = (wave & 1) * 64;

  const int bk   = tid >> 3;           // 0..31
  const int bseg = (tid & 7) * 16;     // 0..112

  v8f acc[2][4] = {};

  for (int kt = 0; kt < Ksize; kt += 32) {
    // ---- A tile (128 x 32 bf16 = 512 x 16B): async global->LDS ----
    {
      const int c0 = tid, c1 = tid + 256;
      const int r0 = c0 >> 2, o0 = (c0 & 3) * 8;
      const int r1 = c1 >> 2, o1 = (c1 & 3) * 8;
      async_cp16(&As[r0 * 40 + o0], Abf + (size_t)(m0 + r0) * Ksize + kt + o0);
      async_cp16(&As[r1 * 40 + o1], Abf + (size_t)(m0 + r1) * Ksize + kt + o1);
    }

    // ---- B tile (32 x 128 f32), convert + transpose -> Bs[n][k] ----
    {
      const float4* wsrc = (const float4*)(W + (size_t)(kt + bk) * Nsize + n0 + bseg);
      float4 g0 = wsrc[0], g1 = wsrc[1], g2 = wsrc[2], g3 = wsrc[3];
      float vals[16] = {g0.x, g0.y, g0.z, g0.w, g1.x, g1.y, g1.z, g1.w,
                        g2.x, g2.y, g2.z, g2.w, g3.x, g3.y, g3.z, g3.w};
#pragma unroll
      for (int i = 0; i < 16; ++i)
        Bs[(bseg + i) * 40 + bk] = (__bf16)vals[i];
      if (kt + 32 < Ksize)
        __builtin_prefetch((const void*)(wsrc + (size_t)8 * Nsize), 0, 3);
    }

    WAIT_ASYNC(0);
    __syncthreads();

    // ---- fragments + 8 WMMA ----
    F16x2 afr[2], bfr[4];
#pragma unroll
    for (int mi = 0; mi < 2; ++mi) {
      const __bf16* base = &As[(wm0 + mi * 16 + ln) * 40 + half * 8];
      afr[mi].h[0] = *(const v8bf*)base;          // K 0-7   (or 8-15)
      afr[mi].h[1] = *(const v8bf*)(base + 16);   // K 16-23 (or 24-31)
    }
#pragma unroll
    for (int ni = 0; ni < 4; ++ni) {
      const __bf16* base = &Bs[(wn0 + ni * 16 + ln) * 40 + half * 8];
      bfr[ni].h[0] = *(const v8bf*)base;          // K 0-7   (or 8-15)
      bfr[ni].h[1] = *(const v8bf*)(base + 16);   // K 16-23 (or 24-31)
    }
#pragma unroll
    for (int mi = 0; mi < 2; ++mi)
#pragma unroll
      for (int ni = 0; ni < 4; ++ni)
        acc[mi][ni] = WMMA_BF16(afr[mi].v, bfr[ni].v, acc[mi][ni]);

    __syncthreads();
  }

  // ---- epilogue: bias + store ----
#pragma unroll
  for (int ni = 0; ni < 4; ++ni) {
    const int gc = n0 + wn0 + ni * 16 + ln;
    const float bv = bias[gc];
    if constexpr (MODE == 0) {
      const int which = gc / DM;          // tile-uniform (768 % 16 == 0)
      const int rem   = gc - which * DM;
      const int h     = rem / DH;
      const int d     = rem - h * DH;
#pragma unroll
      for (int mi = 0; mi < 2; ++mi)
#pragma unroll
        for (int r = 0; r < 8; ++r) {
          const int gm = m0 + wm0 + mi * 16 + r + 8 * half;
          const int b  = gm >> 11;
          const int ns = gm & 2047;
          const int bh = b * NH + h;
          const float val = acc[mi][ni][r] + bv;
          if (which == 0)      qws[((size_t)bh * SEQ + ns) * DH + d]  = (__bf16)val;
          else if (which == 1) kws[((size_t)bh * SEQ + ns) * DH + d]  = (__bf16)val;
          else                 vtws[((size_t)bh * DH + d) * SEQ + ns] = (__bf16)val;
        }
    } else {
#pragma unroll
      for (int mi = 0; mi < 2; ++mi)
#pragma unroll
        for (int r = 0; r < 8; ++r) {
          const int gm = m0 + wm0 + mi * 16 + r + 8 * half;
          outF[(size_t)gm * Nsize + gc] = acc[mi][ni][r] + bv;
        }
    }
  }
}

// ---------------------------------------------------------------------------
// Flash-style attention. Block = 8 waves = 128 query rows of one (b,h).
// K/V tiles for each 32-key chunk are staged once per block into LDS via
// double-buffered async copies (4 async instructions per wave per stage),
// then all waves feed WMMA from LDS.
// ---------------------------------------------------------------------------
constexpr int KS_STRIDE = 104;  // 96 + 8 pad: conflict-free 16-lane access
constexpr int VS_STRIDE = 40;   // 32 + 8 pad

__global__ __launch_bounds__(256)
void attn_wmma(const __bf16* __restrict__ Q, const __bf16* __restrict__ K,
               const __bf16* __restrict__ Vt, __bf16* __restrict__ ctx)
{
  __shared__ __align__(16) __bf16 Ks[2][32 * KS_STRIDE];  // [key][dh]
  __shared__ __align__(16) __bf16 Vs[2][96 * VS_STRIDE];  // [dh][key]
  __shared__ __align__(16) __bf16 Pbuf[8][16][40];        // per-wave P tile

  const int bh   = blockIdx.x >> 4;
  const int qblk = blockIdx.x & 15;
  const int tid  = threadIdx.x;
  const int wave = tid >> 5, lane = tid & 31;
  const int half = lane >> 4, ln = lane & 15;
  const int qw   = qblk * 128 + wave * 16;

  const __bf16* Qb = Q  + (size_t)bh * SEQ * DH;
  const __bf16* Kb = K  + (size_t)bh * SEQ * DH;
  const __bf16* Vb = Vt + (size_t)bh * DH * SEQ;

  // chunk ids for staging: 384 x 16B per K tile and per V tile; make the
  // async-instruction count wave-uniform (4 per stage) by duplicating tails.
  const int ck0 = tid, ck1 = 256 + (tid & 127);

  // Stage K[32 x 96] and Vt-slice[96 x 32] for key chunk j0 into buffer buf.
  auto stage = [&](int buf, int j0) {
    {
      int r = ck0 / 12, cc = (ck0 - r * 12) * 8;       // K: 12 chunks per row
      async_cp16(&Ks[buf][r * KS_STRIDE + cc], Kb + (size_t)(j0 + r) * DH + cc);
      r = ck1 / 12; cc = (ck1 - r * 12) * 8;
      async_cp16(&Ks[buf][r * KS_STRIDE + cc], Kb + (size_t)(j0 + r) * DH + cc);
    }
    {
      int r = ck0 >> 2, cc = (ck0 & 3) * 8;            // V: 4 chunks per row
      async_cp16(&Vs[buf][r * VS_STRIDE + cc], Vb + (size_t)r * SEQ + j0 + cc);
      r = ck1 >> 2; cc = (ck1 & 3) * 8;
      async_cp16(&Vs[buf][r * VS_STRIDE + cc], Vb + (size_t)r * SEQ + j0 + cc);
    }
  };

  // Q fragments: 3 chunks of K-dim (Dh = 96 = 3*32), resident all pass.
  F16x2 qf[3];
#pragma unroll
  for (int d = 0; d < 3; ++d) {
    const __bf16* p = Qb + (size_t)(qw + ln) * DH + d * 32 + half * 8;
    qf[d].h[0] = *(const v8bf*)p;
    qf[d].h[1] = *(const v8bf*)(p + 16);
  }

  float m_run[8], l_run[8];
#pragma unroll
  for (int r = 0; r < 8; ++r) { m_run[r] = -3.0e38f; l_run[r] = 0.f; }
  v8f o[6] = {};

  const float kscale = 0.1020620726159658f;  // 1/sqrt(96)

  stage(0, 0);                                // prologue stage

  for (int j0 = 0; j0 < SEQ; j0 += 32) {
    const int cur = (j0 >> 5) & 1;
    if (j0 + 32 < SEQ) {
      stage(cur ^ 1, j0 + 32);                // overlap next stage with compute
      WAIT_ASYNC(4);                          // current buffer's 4 copies done
    } else {
      WAIT_ASYNC(0);
    }
    __syncthreads();

    // ---- scores for two 16-key subtiles, from LDS ----
    v8f S[2];
#pragma unroll
    for (int s = 0; s < 2; ++s) {
      v8f sc = {};
#pragma unroll
      for (int d = 0; d < 3; ++d) {
        F16x2 kf;
        const __bf16* p = &Ks[cur][(s * 16 + ln) * KS_STRIDE + d * 32 + half * 8];
        kf.h[0] = *(const v8bf*)p;
        kf.h[1] = *(const v8bf*)(p + 16);
        sc = WMMA_BF16(qf[d].v, kf.v, sc);
      }
#pragma unroll
      for (int r = 0; r < 8; ++r) sc[r] *= kscale;
      S[s] = sc;
    }

    // ---- online softmax (row r lives in one VGPR across 16 lanes) ----
    float alpha[8];
#pragma unroll
    for (int r = 0; r < 8; ++r) {
      float mx = fmaxf(S[0][r], S[1][r]);
#pragma unroll
      for (int off = 8; off > 0; off >>= 1)
        mx = fmaxf(mx, __shfl_xor(mx, off, 16));
      const float mn = fmaxf(m_run[r], mx);
      alpha[r] = __expf(m_run[r] - mn);
      m_run[r] = mn;
      const float p0 = __expf(S[0][r] - mn);
      const float p1 = __expf(S[1][r] - mn);
      float sm = p0 + p1;
#pragma unroll
      for (int off = 8; off > 0; off >>= 1)
        sm += __shfl_xor(sm, off, 16);
      l_run[r] = l_run[r] * alpha[r] + sm;
      Pbuf[wave][r + 8 * half][ln]      = (__bf16)p0;   // C-layout -> A-layout
      Pbuf[wave][r + 8 * half][16 + ln] = (__bf16)p1;
    }
#pragma unroll
    for (int t = 0; t < 6; ++t)
#pragma unroll
      for (int r = 0; r < 8; ++r)
        o[t][r] *= alpha[r];

    // ---- P fragment (A layout) from wave-private LDS ----
    F16x2 pf;
    const __bf16* pp = &Pbuf[wave][ln][half * 8];
    pf.h[0] = *(const v8bf*)pp;
    pf.h[1] = *(const v8bf*)(pp + 16);

    // ---- P @ V : 6 Dh tiles from LDS ----
#pragma unroll
    for (int t = 0; t < 6; ++t) {
      F16x2 vf;
      const __bf16* p = &Vs[cur][(t * 16 + ln) * VS_STRIDE + half * 8];
      vf.h[0] = *(const v8bf*)p;
      vf.h[1] = *(const v8bf*)(p + 16);
      o[t] = WMMA_BF16(pf.v, vf.v, o[t]);
    }

    __syncthreads();   // protect buffer reuse before next stage overwrites
  }

  // ---- normalize + store context in [B, N, H*Dh] (GEMM3 A layout) ----
  const int b = bh >> 3, h = bh & 7;
#pragma unroll
  for (int r = 0; r < 8; ++r) {
    const float inv = 1.0f / l_run[r];
    const int gq = qw + r + 8 * half;
    __bf16* crow = ctx + ((size_t)b * SEQ + gq) * DM + h * DH;
#pragma unroll
    for (int t = 0; t < 6; ++t)
      crow[t * 16 + ln] = (__bf16)(o[t][r] * inv);
  }
}

// ---------------------------------------------------------------------------
extern "C" void kernel_launch(void* const* d_in, const int* in_sizes, int n_in,
                              void* d_out, int out_size, void* d_ws, size_t ws_size,
                              hipStream_t stream)
{
  const float* x     = (const float*)d_in[0];
  const float* W_qkv = (const float*)d_in[1];
  const float* b_qkv = (const float*)d_in[2];
  const float* W_out = (const float*)d_in[3];
  const float* b_out = (const float*)d_in[4];
  float* out = (float*)d_out;

  const size_t seg = (size_t)BATCH * NH * SEQ * DH;   // 6,291,456 (== 8192*768)
  __bf16* qws  = (__bf16*)d_ws;
  __bf16* kws  = qws + seg;
  __bf16* vtws = kws + seg;
  __bf16* ctx  = vtws + seg;
  __bf16* xb   = ctx + seg;                            // total ws ~60 MB

  dim3 blk(256);
  const int n8 = (int)(seg / 8);                       // 786432
  cvt_bf16<<<dim3((n8 + 255) / 256), blk, 0, stream>>>(x, xb, n8);
  gemm_wmma<0><<<dim3(QKV_N / 128, M_ROWS / 128), blk, 0, stream>>>(
      xb, W_qkv, b_qkv, nullptr, qws, kws, vtws, DM, QKV_N);
  attn_wmma<<<dim3(BATCH * NH * (SEQ / 128)), blk, 0, stream>>>(qws, kws, vtws, ctx);
  gemm_wmma<1><<<dim3(DM / 128, M_ROWS / 128), blk, 0, stream>>>(
      ctx, W_out, b_out, out, nullptr, nullptr, nullptr, DM, DM);
}
